// DGCNN_PYG_4037269258395
// MI455X (gfx1250) — matrix operand order repaired and checked by reading, not compiled, not run on hardware
//
#include <hip/hip_runtime.h>
#include <hip/hip_bf16.h>

typedef __attribute__((ext_vector_type(16))) _Float16 v16h;
typedef __attribute__((ext_vector_type(8)))  float    v8f;
typedef _Float16 half_t;

#define BGRAPH 64
#define PPTS   1024
#define NPTS   (BGRAPH * PPTS)
#define KNN    5
#define NEDGE  (NPTS * KNN)
#define BN_EPS 1e-5f

// ---------------------------------------------------------------------------
// WMMA helpers (CDNA5 gfx1250, wave32).  Layouts per CDNA5 ISA:
//  A (16x32 f16): lane<16 -> row m=lane, halves[0..7]=K 0..7, halves[8..15]=K 16..23
//                 lane>=16 -> row m=lane-16, K 8..15 and 24..31
//  B (32x16 f16): lane<16 -> col n=lane, halves[0..7]=K 0..7, halves[8..15]=K 16..23
//                 lane>=16 -> col n=lane-16, K 8..15 and 24..31
//  C/D (16x16 f32): vgpr r: lane<16 -> (m=r, n=lane); lane>=16 -> (m=8+r, n=lane-16)
// ---------------------------------------------------------------------------
__device__ __forceinline__ v8f wmma16(v16h a, v16h b, v8f c) {
  return __builtin_amdgcn_wmma_f32_16x16x32_f16(false, a, false, b, (short)0, c,
                                                false, false);
}

// B fragment from LDS *transposed* weights WT[n][k] (row stride ldk halves).
// Tile covers K rows [k0,k0+32), N cols [n0,n0+16).  Per lane: K 8h..8h+7 and
// 16+8h..16+8h+7 (h = lane/16), matching the A fragment order.
__device__ __forceinline__ v16h load_bfragT(const half_t* WT, int ldk, int k0,
                                            int n0, int lane) {
  int n  = n0 + (lane & 15);
  int kb = k0 + ((lane >> 4) << 3);
  const half_t* src = WT + n * ldk + kb;
  v16h b;
#pragma unroll
  for (int h = 0; h < 8; ++h) { b[h] = src[h]; b[8 + h] = src[16 + h]; }
  return b;
}

// A fragment from LDS row-major H[m][k] (row stride ldh halves), K offset k0.
__device__ __forceinline__ v16h load_afrag(const half_t* H, int ldh, int k0,
                                           int lane) {
  int m  = lane & 15;
  int kb = k0 + ((lane >> 4) << 3);
  const half_t* src = H + m * ldh + kb;
  v16h a;
#pragma unroll
  for (int h = 0; h < 8; ++h) a[h] = src[h];
#pragma unroll
  for (int h = 0; h < 8; ++h) a[8 + h] = src[16 + h];
  return a;
}

// Monotone float<->uint encoding so segment-max can use integer atomicMax.
__device__ __forceinline__ unsigned int fenc(float f) {
  unsigned int u = __float_as_uint(f);
  return (u & 0x80000000u) ? ~u : (u | 0x80000000u);
}
__device__ __forceinline__ float fdec(unsigned int e) {
  return __uint_as_float((e & 0x80000000u) ? (e ^ 0x80000000u) : ~e);
}

// ---------------------------------------------------------------------------
// kNN (exact fp32 distances; stable top-5 to match jax.lax.top_k tie order)
// grid = BGRAPH*8 blocks of 128 threads; 1 thread per query row.
// ---------------------------------------------------------------------------
template <int D>
__global__ void __launch_bounds__(128) knn_kernel(const float* __restrict__ x,
                                                  int* __restrict__ nidx) {
  __shared__ float tile[128 * D];
  const int cloud = blockIdx.x >> 3;
  const int row   = ((blockIdx.x & 7) << 7) + threadIdx.x;
  const int gRow  = cloud * PPTS + row;

  float xr[D];
#pragma unroll
  for (int d = 0; d < D; ++d) xr[d] = x[gRow * D + d];

  float bd[KNN];
  int   bi[KNN];
#pragma unroll
  for (int k = 0; k < KNN; ++k) { bd[k] = 3.0e38f; bi[k] = 0; }

  for (int tb = 0; tb < PPTS; tb += 128) {
    __syncthreads();
    for (int i = threadIdx.x; i < 128 * D; i += 128)
      tile[i] = x[(cloud * PPTS + tb) * D + i];
    __syncthreads();
    for (int c = 0; c < 128; ++c) {
      float d2 = 0.f;
#pragma unroll
      for (int d = 0; d < D; ++d) {
        float t = xr[d] - tile[c * D + d];
        d2 = fmaf(t, t, d2);
      }
      if (d2 < bd[KNN - 1]) {           // strict <  => earliest index wins ties
        bd[KNN - 1] = d2;
        bi[KNN - 1] = cloud * PPTS + tb + c;
#pragma unroll
        for (int k = KNN - 1; k > 0; --k) {
          if (bd[k] < bd[k - 1]) {
            float td = bd[k]; bd[k] = bd[k - 1]; bd[k - 1] = td;
            int   ti = bi[k]; bi[k] = bi[k - 1]; bi[k - 1] = ti;
          }
        }
      }
    }
  }
#pragma unroll
  for (int k = 0; k < KNN; ++k) nidx[gRow * KNN + k] = bi[k];
}

// ---------------------------------------------------------------------------
// conv1: fused 3-layer edge MLP 12->64->64->64 (BN folded) + max over K.
// Block = 160 threads = 5 waves; each wave owns a 16-edge tile; block = 16 pts.
// ---------------------------------------------------------------------------
__global__ void __launch_bounds__(160) conv1_kernel(
    const float* __restrict__ pos, const int* __restrict__ nidx,
    const float* __restrict__ w1, const float* __restrict__ b1,
    const float* __restrict__ g1, const float* __restrict__ be1,
    const float* __restrict__ w2, const float* __restrict__ b2,
    const float* __restrict__ g2, const float* __restrict__ be2,
    const float* __restrict__ w3, const float* __restrict__ b3,
    float* __restrict__ x1f, half_t* __restrict__ x1h) {
  __shared__ half_t W1T[64 * 40];        // [n][k], K padded 12->32 with zeros
  __shared__ half_t W2T[64 * 72];        // [n][k]
  __shared__ half_t W3T[64 * 72];
  __shared__ float  sc1[64], bc1[64], sc2[64], bc2[64], bc3[64];
  __shared__ half_t Hst[5][16 * 72];     // per-wave activation staging
  __shared__ half_t hbuf[80][72];        // per-edge layer-3 output

  const int tid = threadIdx.x;
  if (tid < 64) {
    float s1 = g1[tid] * rsqrtf(1.f + BN_EPS);
    sc1[tid] = s1; bc1[tid] = b1[tid] * s1 + be1[tid];
    float s2 = g2[tid] * rsqrtf(1.f + BN_EPS);
    sc2[tid] = s2; bc2[tid] = b2[tid] * s2 + be2[tid];
    bc3[tid] = b3[tid];
  }
  for (int i = tid; i < 64 * 32; i += 160) {
    int n = i >> 5, k = i & 31;
    W1T[n * 40 + k] = (k < 12) ? (half_t)w1[k * 64 + n] : (half_t)0.f;
  }
  for (int i = tid; i < 64 * 64; i += 160) {
    int n = i >> 6, k = i & 63;
    W2T[n * 72 + k] = (half_t)w2[k * 64 + n];
    W3T[n * 72 + k] = (half_t)w3[k * 64 + n];
  }
  __syncthreads();

  const int wave  = tid >> 5, lane = tid & 31;
  const int pBase = blockIdx.x << 4;
  const int m     = lane & 15;
  const int eg    = pBase * KNN + wave * 16 + m;
  const int p     = eg / KNN;
  const int j     = nidx[eg];
  const int mb    = (lane >> 4) << 3;
  const int nC    = lane & 15;

  // layer-1 A fragment: feats = [x_i(6), x_j-x_i(6), 0-pad to 32]
  v16h a1;
  {
    float xi[6], xj[6];
#pragma unroll
    for (int d = 0; d < 6; ++d) { xi[d] = pos[p * 6 + d]; xj[d] = pos[j * 6 + d]; }
#pragma unroll
    for (int h = 0; h < 16; ++h) a1[h] = (half_t)0.f;
    if (lane < 16) {   // K = 0..7  (xi0..5, d0, d1)
      a1[0] = (half_t)xi[0]; a1[1] = (half_t)xi[1]; a1[2] = (half_t)xi[2];
      a1[3] = (half_t)xi[3]; a1[4] = (half_t)xi[4]; a1[5] = (half_t)xi[5];
      a1[6] = (half_t)(xj[0] - xi[0]); a1[7] = (half_t)(xj[1] - xi[1]);
    } else {           // K = 8..15 (d2..d5, zeros)
      a1[0] = (half_t)(xj[2] - xi[2]); a1[1] = (half_t)(xj[3] - xi[3]);
      a1[2] = (half_t)(xj[4] - xi[4]); a1[3] = (half_t)(xj[5] - xi[5]);
    }
  }

  half_t* Hw = Hst[wave];

  // layer 1: 12(->32) -> 64, BN+ReLU
#pragma unroll
  for (int nt = 0; nt < 4; ++nt) {
    v8f c = {};
    c = wmma16(a1, load_bfragT(W1T, 40, 0, nt * 16, lane), c);
    int n = nt * 16 + nC;
    float s = sc1[n], bb = bc1[n];
#pragma unroll
    for (int r = 0; r < 8; ++r)
      Hw[(mb + r) * 72 + n] = (half_t)fmaxf(c[r] * s + bb, 0.f);
  }
  // layer 2: 64 -> 64, BN+ReLU
  {
    v16h a0 = load_afrag(Hw, 72, 0, lane);
    v16h aH = load_afrag(Hw, 72, 32, lane);
#pragma unroll
    for (int nt = 0; nt < 4; ++nt) {
      v8f c = {};
      c = wmma16(a0, load_bfragT(W2T, 72, 0,  nt * 16, lane), c);
      c = wmma16(aH, load_bfragT(W2T, 72, 32, nt * 16, lane), c);
      int n = nt * 16 + nC;
      float s = sc2[n], bb = bc2[n];
#pragma unroll
      for (int r = 0; r < 8; ++r)
        Hw[(mb + r) * 72 + n] = (half_t)fmaxf(c[r] * s + bb, 0.f);
    }
  }
  // layer 3: 64 -> 64, plain
  {
    v16h a0 = load_afrag(Hw, 72, 0, lane);
    v16h aH = load_afrag(Hw, 72, 32, lane);
#pragma unroll
    for (int nt = 0; nt < 4; ++nt) {
      v8f c = {};
      c = wmma16(a0, load_bfragT(W3T, 72, 0,  nt * 16, lane), c);
      c = wmma16(aH, load_bfragT(W3T, 72, 32, nt * 16, lane), c);
      int n = nt * 16 + nC;
      float bb = bc3[n];
#pragma unroll
      for (int r = 0; r < 8; ++r)
        hbuf[wave * 16 + mb + r][n] = (half_t)(c[r] + bb);
    }
  }
  __syncthreads();
  // max over K=5 neighbors
  for (int v = tid; v < 16 * 64; v += 160) {
    int pl = v >> 6, f = v & 63;
    float mx = (float)hbuf[pl * KNN + 0][f];
#pragma unroll
    for (int k = 1; k < KNN; ++k) mx = fmaxf(mx, (float)hbuf[pl * KNN + k][f]);
    int gp = pBase + pl;
    x1f[gp * 64 + f] = mx;
    x1h[gp * 64 + f] = (half_t)mx;
  }
}

// ---------------------------------------------------------------------------
// conv2: edge Linear 128->128 + max over K.  Same 5-wave / 16-point blocks.
// ---------------------------------------------------------------------------
__global__ void __launch_bounds__(160) conv2_kernel(
    const half_t* __restrict__ x1h, const int* __restrict__ nidx,
    const float* __restrict__ w, const float* __restrict__ b,
    half_t* __restrict__ x2h) {
  __shared__ half_t WT[128 * 136];       // [n][k]
  __shared__ float  bc[128];
  __shared__ half_t hbuf[80][136];

  const int tid = threadIdx.x;
  if (tid < 128) bc[tid] = b[tid];
  for (int i = tid; i < 128 * 128; i += 160) {
    int n = i >> 7, k = i & 127;
    WT[n * 136 + k] = (half_t)w[k * 128 + n];
  }
  __syncthreads();

  const int wave  = tid >> 5, lane = tid & 31;
  const int pBase = blockIdx.x << 4;
  const int m     = lane & 15;
  const int eg    = pBase * KNN + wave * 16 + m;
  const int p     = eg / KNN;
  const int j     = nidx[eg];
  const int kb    = (lane >> 4) << 3;
  const int mb    = (lane >> 4) << 3;
  const int nC    = lane & 15;

  // A frags: ksteps 0,1 = x1_i (feats 0..63); ksteps 2,3 = x1_j - x1_i
  v16h A[4];
#pragma unroll
  for (int s = 0; s < 2; ++s) {
    const half_t* xp = x1h + p * 64 + s * 32 + kb;
#pragma unroll
    for (int h = 0; h < 8; ++h) { A[s][h] = xp[h]; A[s][8 + h] = xp[16 + h]; }
  }
#pragma unroll
  for (int s = 0; s < 2; ++s) {
    const half_t* xp = x1h + p * 64 + s * 32 + kb;
    const half_t* xq = x1h + j * 64 + s * 32 + kb;
#pragma unroll
    for (int h = 0; h < 8; ++h) {
      A[2 + s][h]     = (half_t)(xq[h] - xp[h]);
      A[2 + s][8 + h] = (half_t)(xq[16 + h] - xp[16 + h]);
    }
  }

#pragma unroll
  for (int nt = 0; nt < 8; ++nt) {
    v8f c = {};
#pragma unroll
    for (int s = 0; s < 4; ++s)
      c = wmma16(A[s], load_bfragT(WT, 136, s * 32, nt * 16, lane), c);
    int n = nt * 16 + nC;
    float bb = bc[n];
#pragma unroll
    for (int r = 0; r < 8; ++r)
      hbuf[wave * 16 + mb + r][n] = (half_t)(c[r] + bb);
  }
  __syncthreads();
  for (int v = tid; v < 16 * 128; v += 160) {
    int pl = v >> 7, f = v & 127;
    float mx = (float)hbuf[pl * KNN + 0][f];
#pragma unroll
    for (int k = 1; k < KNN; ++k) mx = fmaxf(mx, (float)hbuf[pl * KNN + k][f]);
    x2h[(pBase + pl) * 128 + f] = (half_t)mx;
  }
}

// ---------------------------------------------------------------------------
// lin1 + fused per-cloud max-pool.  Block = 256 thr (8 waves) = 128 points of
// one cloud; loops over eight 128-wide N chunks of the 192x1024 weight.
// ---------------------------------------------------------------------------
__global__ void __launch_bounds__(256) lin1_pool_kernel(
    const half_t* __restrict__ x1h, const half_t* __restrict__ x2h,
    const float* __restrict__ w, const float* __restrict__ b,
    unsigned int* __restrict__ pooled) {
  __shared__ half_t WT[128 * 200];       // [n][k], k < 192
  __shared__ float  colmax[8][128];

  const int tid   = threadIdx.x;
  const int cloud = blockIdx.x >> 3;
  const int chunkP = blockIdx.x & 7;
  const int pBase = cloud * PPTS + chunkP * 128;
  const int wave  = tid >> 5, lane = tid & 31;
  const int m     = lane & 15;
  const int p     = pBase + wave * 16 + m;
  const int kb    = (lane >> 4) << 3;

  // A frags: ksteps 0,1 from x1 (64), ksteps 2..5 from x2 (128)
  v16h A[6];
#pragma unroll
  for (int s = 0; s < 2; ++s) {
    const half_t* xp = x1h + p * 64 + s * 32 + kb;
#pragma unroll
    for (int h = 0; h < 8; ++h) { A[s][h] = xp[h]; A[s][8 + h] = xp[16 + h]; }
  }
#pragma unroll
  for (int s = 0; s < 4; ++s) {
    const half_t* xp = x2h + p * 128 + s * 32 + kb;
#pragma unroll
    for (int h = 0; h < 8; ++h) { A[2 + s][h] = xp[h]; A[2 + s][8 + h] = xp[16 + h]; }
  }

  for (int nc = 0; nc < 8; ++nc) {
    for (int i = tid; i < 192 * 128; i += 256) {
      int k = i >> 7, n = i & 127;
      WT[n * 200 + k] = (half_t)w[k * 1024 + nc * 128 + n];
    }
    __syncthreads();
#pragma unroll
    for (int nt = 0; nt < 8; ++nt) {
      v8f c = {};
#pragma unroll
      for (int s = 0; s < 6; ++s)
        c = wmma16(A[s], load_bfragT(WT, 200, s * 32, nt * 16, lane), c);
      float rm = c[0];
#pragma unroll
      for (int r = 1; r < 8; ++r) rm = fmaxf(rm, c[r]);
      rm = fmaxf(rm, __shfl_xor(rm, 16, 32));   // combine rows 0..7 with 8..15
      if (lane < 16) colmax[wave][nt * 16 + lane] = rm;
    }
    __syncthreads();
    if (tid < 128) {
      float mx = colmax[0][tid];
#pragma unroll
      for (int wv = 1; wv < 8; ++wv) mx = fmaxf(mx, colmax[wv][tid]);
      mx += b[nc * 128 + tid];                  // bias uniform per column
      atomicMax(&pooled[cloud * 1024 + nc * 128 + tid], fenc(mx));
    }
    __syncthreads();
  }
}

// ---------------------------------------------------------------------------
// head: MLP 1024->512->256->512 + log_softmax.  One block per graph.
// ---------------------------------------------------------------------------
__global__ void __launch_bounds__(256) head_kernel(
    const unsigned int* __restrict__ pooled,
    const float* __restrict__ w1, const float* __restrict__ b1,
    const float* __restrict__ w2, const float* __restrict__ b2,
    const float* __restrict__ w3, const float* __restrict__ b3,
    float* __restrict__ out) {
  __shared__ float xin[1024], h1[512], h2[256], h3[512], red[256];
  const int g = blockIdx.x, tid = threadIdx.x;

  for (int i = tid; i < 1024; i += 256) xin[i] = fdec(pooled[g * 1024 + i]);
  __syncthreads();
  for (int f = tid; f < 512; f += 256) {
    float acc = b1[f];
    for (int k = 0; k < 1024; ++k) acc = fmaf(xin[k], w1[k * 512 + f], acc);
    h1[f] = fmaxf(acc, 0.f);
  }
  __syncthreads();
  {
    float acc = b2[tid];
    for (int k = 0; k < 512; ++k) acc = fmaf(h1[k], w2[k * 256 + tid], acc);
    h2[tid] = fmaxf(acc, 0.f);
  }
  __syncthreads();
  for (int f = tid; f < 512; f += 256) {
    float acc = b3[f];
    for (int k = 0; k < 256; ++k) acc = fmaf(h2[k], w3[k * 512 + f], acc);
    h3[f] = acc;
  }
  __syncthreads();
  float lm = -3.0e38f;
  for (int i = tid; i < 512; i += 256) lm = fmaxf(lm, h3[i]);
  red[tid] = lm;
  __syncthreads();
  for (int s = 128; s > 0; s >>= 1) {
    if (tid < s) red[tid] = fmaxf(red[tid], red[tid + s]);
    __syncthreads();
  }
  float mx = red[0];
  __syncthreads();
  float ls = 0.f;
  for (int i = tid; i < 512; i += 256) ls += expf(h3[i] - mx);
  red[tid] = ls;
  __syncthreads();
  for (int s = 128; s > 0; s >>= 1) {
    if (tid < s) red[tid] += red[tid + s];
    __syncthreads();
  }
  float lse = mx + logf(red[0]);
  __syncthreads();
  for (int i = tid; i < 512; i += 256) out[g * 512 + i] = h3[i] - lse;
}

// ---------------------------------------------------------------------------
extern "C" void kernel_launch(void* const* d_in, const int* in_sizes, int n_in,
                              void* d_out, int out_size, void* d_ws,
                              size_t ws_size, hipStream_t stream) {
  (void)in_sizes; (void)n_in; (void)out_size; (void)ws_size;
  const float* pos   = (const float*)d_in[0];
  const float* c1w1  = (const float*)d_in[3];
  const float* c1b1  = (const float*)d_in[4];
  const float* c1g1  = (const float*)d_in[5];
  const float* c1be1 = (const float*)d_in[6];
  const float* c1w2  = (const float*)d_in[7];
  const float* c1b2  = (const float*)d_in[8];
  const float* c1g2  = (const float*)d_in[9];
  const float* c1be2 = (const float*)d_in[10];
  const float* c1w3  = (const float*)d_in[11];
  const float* c1b3  = (const float*)d_in[12];
  const float* c2w   = (const float*)d_in[13];
  const float* c2b   = (const float*)d_in[14];
  const float* l1w   = (const float*)d_in[15];
  const float* l1b   = (const float*)d_in[16];
  const float* mw1   = (const float*)d_in[17];
  const float* mb1   = (const float*)d_in[18];
  const float* mw2   = (const float*)d_in[19];
  const float* mb2   = (const float*)d_in[20];
  const float* mw3   = (const float*)d_in[21];
  const float* mb3   = (const float*)d_in[22];

  char* ws = (char*)d_ws;
  size_t off = 0;
  auto carve = [&](size_t bytes) -> void* {
    void* pp = ws + off;
    off += (bytes + 255) & ~(size_t)255;
    return pp;
  };
  int*          idx1   = (int*)carve((size_t)NEDGE * 4);
  int*          idx2   = (int*)carve((size_t)NEDGE * 4);
  float*        x1f    = (float*)carve((size_t)NPTS * 64 * 4);
  half_t*       x1h    = (half_t*)carve((size_t)NPTS * 64 * 2);
  half_t*       x2h    = (half_t*)carve((size_t)NPTS * 128 * 2);
  unsigned int* pooled = (unsigned int*)carve((size_t)BGRAPH * 1024 * 4);

  knn_kernel<6><<<BGRAPH * 8, 128, 0, stream>>>(pos, idx1);
  conv1_kernel<<<NPTS / 16, 160, 0, stream>>>(pos, idx1, c1w1, c1b1, c1g1, c1be1,
                                              c1w2, c1b2, c1g2, c1be2, c1w3,
                                              c1b3, x1f, x1h);
  knn_kernel<64><<<BGRAPH * 8, 128, 0, stream>>>(x1f, idx2);
  conv2_kernel<<<NPTS / 16, 160, 0, stream>>>(x1h, idx2, c2w, c2b, x2h);
  hipMemsetAsync(pooled, 0, (size_t)BGRAPH * 1024 * 4, stream);
  lin1_pool_kernel<<<BGRAPH * 8, 256, 0, stream>>>(x1h, x2h, l1w, l1b, pooled);
  head_kernel<<<BGRAPH, 256, 0, stream>>>(pooled, mw1, mb1, mw2, mb2, mw3, mb3,
                                          (float*)d_out);
}
